// GPT2Attention_49624052138686
// MI455X (gfx1250) — matrix operand order repaired and checked by reading, not compiled, not run on hardware
//
#include <hip/hip_runtime.h>
#include <hip/hip_bf16.h>

typedef __attribute__((ext_vector_type(16))) _Float16 v16h;
typedef __attribute__((ext_vector_type(8)))  _Float16 v8h;
typedef __attribute__((ext_vector_type(8)))  float    v8f;

namespace {
constexpr int kB  = 2;
constexpr int kT  = 2048;
constexpr int kC  = 1024;
constexpr int kH  = 16;
constexpr int kD  = 64;          // head dim
constexpr int kM  = kB * kT;     // 4096 token rows
constexpr int kNq = 3 * kC;      // 3072 qkv columns
}

// Build a 16-half A/B fragment from two contiguous 16-byte chunks.
static __device__ __forceinline__ v16h ldfrag(const _Float16* p0, const _Float16* p1) {
  v8h lo = *reinterpret_cast<const v8h*>(p0);
  v8h hi = *reinterpret_cast<const v8h*>(p1);
  return __builtin_shufflevector(lo, hi, 0,1,2,3,4,5,6,7,8,9,10,11,12,13,14,15);
}

static __device__ __forceinline__ v8f wmma16(v16h a, v16h b, v8f c) {
  // (neg_a, A, neg_b, B, c_mod, C, reuse_a, reuse_b)
  return __builtin_amdgcn_wmma_f32_16x16x32_f16(false, a, false, b, (short)0, c, false, false);
}

// 64x64 register-blocked GEMM core: 4x4 grid of 16x16x32 WMMAs per k-step.
// Needs ~220 VGPRs -> callers must be compiled with a relaxed waves-per-EU bound.
static __device__ __forceinline__ void gemm_core64(const _Float16* __restrict__ Abase,
                                                   const _Float16* __restrict__ Bbase,
                                                   int K, int kb0, int kbB,
                                                   v8f acc[4][4]) {
#pragma unroll 2
  for (int k = 0; k < K; k += 32) {
    v16h a[4], b[4];
#pragma unroll
    for (int i = 0; i < 4; ++i) {
      const _Float16* ap = Abase + (size_t)i * 16 * K + k;
      a[i] = ldfrag(ap + kb0, ap + kb0 + 16);
      const _Float16* bp = Bbase + (size_t)i * 16 * K + k;
      b[i] = ldfrag(bp + kbB, bp + kbB + 16);
      // 1KB lookahead into the next k-panels (global_prefetch_b8; lanes cover all 16 rows)
      __builtin_prefetch(ap + 512, 0, 1);
      __builtin_prefetch(bp + 512, 0, 1);
    }
#pragma unroll
    for (int i = 0; i < 4; ++i)
#pragma unroll
      for (int j = 0; j < 4; ++j)
        acc[i][j] = wmma16(a[i], b[j], acc[i][j]);
  }
}

// ---------------- conversion kernels ----------------

__global__ void cvt_f32_to_f16_kernel(const float* __restrict__ src,
                                      _Float16* __restrict__ dst, int n) {
  int i = blockIdx.x * blockDim.x + threadIdx.x;
  if (i < n) dst[i] = (_Float16)src[i];
}

// src [R,S] f32 -> dst [S,R] f16   (weights stored N-major so B-fragments are contiguous)
__global__ void transpose_cvt_kernel(const float* __restrict__ src,
                                     _Float16* __restrict__ dst, int R, int S) {
  int i = blockIdx.x * blockDim.x + threadIdx.x;
  if (i < R * S) {
    int n = i / R;      // output row (original column)
    int k = i % R;      // output col (original row)
    dst[i] = (_Float16)src[(size_t)k * S + n];
  }
}

// ---------------- QKV GEMM:  qkv = x @ W_qkv + b, scattered into Q,K,Vt ----------------
// One wave per 64x64 output block; waves-per-eu=1 so the 4x4 f32 accumulator
// block (128 VGPRs) + fragments stay entirely in registers (no scratch spills).

__global__ void __launch_bounds__(256, 1)
qkv_gemm_kernel(const _Float16* __restrict__ x16,   // [kM, kC]
                const _Float16* __restrict__ wT,    // [kNq, kC]
                const float*    __restrict__ bias,  // [kNq]
                _Float16* __restrict__ Qb,          // [B,H,T,D] pre-scaled by 1/8
                _Float16* __restrict__ Kb,          // [B,H,T,D]
                _Float16* __restrict__ Vt) {        // [B,H,D,T]
  const int lane = threadIdx.x & 31;
  const int wave = (blockIdx.x * (blockDim.x >> 5)) + (threadIdx.x >> 5);
  const int ntn  = kNq / 64;                 // 48 column blocks
  const int tm   = wave / ntn;
  const int tn   = wave % ntn;
  const int l15  = lane & 15;
  const int kb0  = (lane < 16) ? 0 : 8;      // A fragment K base
  const int kbB  = (lane < 16) ? 0 : 8;      // B fragment K base
  const int rbase = (lane < 16) ? 0 : 8;

  const _Float16* Abase = x16 + (size_t)(tm * 64 + l15) * kC;
  const _Float16* Bbase = wT  + (size_t)(tn * 64 + l15) * kC;

  v8f acc[4][4] = {};
  gemm_core64(Abase, Bbase, kC, kb0, kbB, acc);

#pragma unroll
  for (int j = 0; j < 4; ++j) {
    const int ncol = tn * 64 + j * 16 + l15;
    const float bn = bias[ncol];
    const int qkv_id = ncol >> 10;           // 0=q 1=k 2=v
    const int c = ncol & 1023;
    const int h = c >> 6;
    const int d = c & 63;
#pragma unroll
    for (int i = 0; i < 4; ++i) {
#pragma unroll
      for (int r = 0; r < 8; ++r) {
        const int m = tm * 64 + i * 16 + rbase + r;
        const int b_ = m >> 11, t = m & (kT - 1);   // kT = 2048
        const size_t bh = (size_t)b_ * kH + h;
        const float v = acc[i][j][r] + bn;
        if (qkv_id == 0)      Qb[(bh * kT + t) * kD + d] = (_Float16)(v * 0.125f);
        else if (qkv_id == 1) Kb[(bh * kT + t) * kD + d] = (_Float16)v;
        else                  Vt[(bh * kD + d) * kT + t] = (_Float16)v;
      }
    }
  }
}

// ---------------- flash attention: one wave per (b,h, 16-row q tile) ----------------

__global__ void attn_kernel(const _Float16* __restrict__ Qb,
                            const _Float16* __restrict__ Kb,
                            const _Float16* __restrict__ Vt,
                            _Float16* __restrict__ attn16) {   // [kM, kC]
  __shared__ __align__(16) _Float16 Plds[8 * 16 * 32];   // per-wave 16x32 P tile

  const int lane = threadIdx.x & 31;
  const int wave = threadIdx.x >> 5;
  const int qtile = blockIdx.x * (blockDim.x >> 5) + wave;
  const int tiles_per_head = kT / 16;
  const int bh = qtile / tiles_per_head;
  const int qt = qtile % tiles_per_head;
  const int qbase = qt * 16;

  const _Float16* Qrow = Qb + (size_t)bh * kT * kD;
  const _Float16* Krow = Kb + (size_t)bh * kT * kD;
  const _Float16* Vrow = Vt + (size_t)bh * kD * kT;
  _Float16* P = &Plds[wave * 16 * 32];

  const int l15 = lane & 15;
  const int kb0 = (lane < 16) ? 0 : 8;
  const int kbB = (lane < 16) ? 0 : 8;
  const int rbase = (lane < 16) ? 0 : 8;

  // Q A-fragments for d=0..31 and d=32..63 (loaded once)
  const _Float16* qp = Qrow + (size_t)(qbase + l15) * kD;
  const v16h aq0 = ldfrag(qp + kb0,      qp + kb0 + 16);
  const v16h aq1 = ldfrag(qp + 32 + kb0, qp + 32 + kb0 + 16);

  v8f o0 = {}, o1 = {}, o2 = {}, o3 = {};
  float mrow[8], lrow[8];
#pragma unroll
  for (int r = 0; r < 8; ++r) { mrow[r] = -3.0e38f; lrow[r] = 0.0f; }

  const int nblk = ((qbase + 15) >> 5) + 1;   // causal: kv blocks with kv_base <= q_hi
  for (int blk = 0; blk < nblk; ++blk) {
    const int kvb = blk * 32;

    // S = Q K^T for two 16-col tiles of this 32-kv block
    v8f s0 = {}, s1 = {};
    {
      const _Float16* kp0 = Krow + (size_t)(kvb + l15) * kD;
      s0 = wmma16(aq0, ldfrag(kp0 + kbB,      kp0 + kbB + 16),      s0);
      s0 = wmma16(aq1, ldfrag(kp0 + 32 + kbB, kp0 + 32 + kbB + 16), s0);
      const _Float16* kp1 = Krow + (size_t)(kvb + 16 + l15) * kD;
      s1 = wmma16(aq0, ldfrag(kp1 + kbB,      kp1 + kbB + 16),      s1);
      s1 = wmma16(aq1, ldfrag(kp1 + 32 + kbB, kp1 + 32 + kbB + 16), s1);
    }

    // causal mask + online softmax (row = one VGPR across a 16-lane half-group)
    const int kv0 = kvb + l15, kv1 = kvb + 16 + l15;
    float p0[8], p1[8];
#pragma unroll
    for (int r = 0; r < 8; ++r) {
      const int q = qbase + rbase + r;
      float a = (kv0 <= q) ? s0[r] : -1.0e30f;
      float c = (kv1 <= q) ? s1[r] : -1.0e30f;
      float mx = fmaxf(a, c);
      for (int off = 8; off >= 1; off >>= 1) mx = fmaxf(mx, __shfl_xor(mx, off, 32));
      const float mnew  = fmaxf(mrow[r], mx);
      const float scale = __expf(mrow[r] - mnew);
      const float e0 = __expf(a - mnew);
      const float e1 = __expf(c - mnew);
      float rs = e0 + e1;
      for (int off = 8; off >= 1; off >>= 1) rs += __shfl_xor(rs, off, 32);
      lrow[r] = lrow[r] * scale + rs;
      mrow[r] = mnew;
      o0[r] *= scale; o1[r] *= scale; o2[r] *= scale; o3[r] *= scale;
      p0[r] = e0; p1[r] = e1;
    }

    // C-layout -> A-layout transpose of P through LDS (per-wave region, DS is in-order)
#pragma unroll
    for (int r = 0; r < 8; ++r) {
      P[(rbase + r) * 32 + l15]      = (_Float16)p0[r];
      P[(rbase + r) * 32 + 16 + l15] = (_Float16)p1[r];
    }
    asm volatile("s_wait_dscnt 0" ::: "memory");

    const _Float16* pm = P + l15 * 32;
    const v16h ap = ldfrag(pm + kb0, pm + kb0 + 16);

    // O += P @ V  (Vt layout makes B-fragments contiguous along kv)
    const size_t vcol = (size_t)kvb + kbB;
    const _Float16* vp0 = Vrow + (size_t)(l15)      * kT + vcol;
    const _Float16* vp1 = Vrow + (size_t)(16 + l15) * kT + vcol;
    const _Float16* vp2 = Vrow + (size_t)(32 + l15) * kT + vcol;
    const _Float16* vp3 = Vrow + (size_t)(48 + l15) * kT + vcol;
    o0 = wmma16(ap, ldfrag(vp0, vp0 + 16), o0);
    o1 = wmma16(ap, ldfrag(vp1, vp1 + 16), o1);
    o2 = wmma16(ap, ldfrag(vp2, vp2 + 16), o2);
    o3 = wmma16(ap, ldfrag(vp3, vp3 + 16), o3);
    asm volatile("s_wait_dscnt 0" ::: "memory");   // P reads done before next block's stores
  }

  // normalize and store to [B,T,C] f16 for the projection GEMM
  const int b_ = bh / kH, h = bh % kH;
#pragma unroll
  for (int r = 0; r < 8; ++r) {
    const int t = qbase + rbase + r;
    const size_t rowoff = ((size_t)b_ * kT + t) * kC + h * kD;
    const float inv = 1.0f / lrow[r];
    attn16[rowoff +  0 + l15] = (_Float16)(o0[r] * inv);
    attn16[rowoff + 16 + l15] = (_Float16)(o1[r] * inv);
    attn16[rowoff + 32 + l15] = (_Float16)(o2[r] * inv);
    attn16[rowoff + 48 + l15] = (_Float16)(o3[r] * inv);
  }
}

// ---------------- projection GEMM: out = attn @ W_proj + b_proj (f32 out) ----------------
// One wave per 64x64 output block.

__global__ void __launch_bounds__(256, 1)
proj_gemm_kernel(const _Float16* __restrict__ a16,  // [kM, kC]
                 const _Float16* __restrict__ wT,   // [kC, kC]
                 const float*    __restrict__ bias, // [kC]
                 float* __restrict__ out) {         // [kM, kC]
  const int lane = threadIdx.x & 31;
  const int wave = (blockIdx.x * (blockDim.x >> 5)) + (threadIdx.x >> 5);
  const int ntn  = kC / 64;                  // 16 column blocks
  const int tm   = wave / ntn;
  const int tn   = wave % ntn;
  const int l15  = lane & 15;
  const int kb0  = (lane < 16) ? 0 : 8;
  const int kbB  = (lane < 16) ? 0 : 8;
  const int rbase = (lane < 16) ? 0 : 8;

  const _Float16* Abase = a16 + (size_t)(tm * 64 + l15) * kC;
  const _Float16* Bbase = wT  + (size_t)(tn * 64 + l15) * kC;

  v8f acc[4][4] = {};
  gemm_core64(Abase, Bbase, kC, kb0, kbB, acc);

#pragma unroll
  for (int j = 0; j < 4; ++j) {
    const int ncol = tn * 64 + j * 16 + l15;
    const float bn = bias[ncol];
#pragma unroll
    for (int i = 0; i < 4; ++i) {
#pragma unroll
      for (int r = 0; r < 8; ++r) {
        const int m = tm * 64 + i * 16 + rbase + r;
        out[(size_t)m * kC + ncol] = acc[i][j][r] + bn;
      }
    }
  }
}

// ---------------- launch ----------------

extern "C" void kernel_launch(void* const* d_in, const int* in_sizes, int n_in,
                              void* d_out, int out_size, void* d_ws, size_t ws_size,
                              hipStream_t stream) {
  (void)in_sizes; (void)n_in; (void)out_size; (void)ws_size;
  const float* x      = (const float*)d_in[0];
  const float* W_qkv  = (const float*)d_in[1];
  const float* b_qkv  = (const float*)d_in[2];
  const float* W_proj = (const float*)d_in[3];
  const float* b_proj = (const float*)d_in[4];
  float* out = (float*)d_out;

  // workspace carve-up (all f16), ~48 MiB total
  char* ws = (char*)d_ws;
  size_t off = 0;
  auto take = [&](size_t elems) { _Float16* p = (_Float16*)(ws + off); off += elems * sizeof(_Float16); return p; };
  _Float16* x16    = take((size_t)kM * kC);      // 8 MiB
  _Float16* wqkvT  = take((size_t)kNq * kC);     // 6 MiB
  _Float16* wprojT = take((size_t)kC * kC);      // 2 MiB
  _Float16* Qb     = take((size_t)kM * kC);      // 8 MiB
  _Float16* Kb     = take((size_t)kM * kC);      // 8 MiB
  _Float16* Vt     = take((size_t)kM * kC);      // 8 MiB
  _Float16* attn16 = take((size_t)kM * kC);      // 8 MiB

  // 1) precision conversion / weight transpose
  {
    int n = kM * kC;
    cvt_f32_to_f16_kernel<<<dim3(n / 256), dim3(256), 0, stream>>>(x, x16, n);
  }
  {
    int n = kC * kNq;
    transpose_cvt_kernel<<<dim3(n / 256), dim3(256), 0, stream>>>(W_qkv, wqkvT, kC, kNq);
  }
  {
    int n = kC * kC;
    transpose_cvt_kernel<<<dim3(n / 256), dim3(256), 0, stream>>>(W_proj, wprojT, kC, kC);
  }

  // 2) QKV GEMM: (4096/64)*(3072/64) = 3072 waves, 8 waves/block
  qkv_gemm_kernel<<<dim3(3072 / 8), dim3(256), 0, stream>>>(x16, wqkvT, b_qkv, Qb, Kb, Vt);

  // 3) flash attention: B*H*(T/16) = 4096 q-tiles, 8 waves/block
  attn_kernel<<<dim3(4096 / 8), dim3(256), 0, stream>>>(Qb, Kb, Vt, attn16);

  // 4) output projection: (4096/64)*(1024/64) = 1024 waves, 8 waves/block
  proj_gemm_kernel<<<dim3(1024 / 8), dim3(256), 0, stream>>>(attn16, wprojT, b_proj, out);
}
